// SelfAttentionHead_5368709120783
// MI455X (gfx1250) — matrix operand order repaired and checked by reading, not compiled, not run on hardware
//
#include <hip/hip_runtime.h>
#include <hip/hip_bf16.h>
#include <stdint.h>

#define BB 16
#define TT 2048
#define CC 1024
#define HH 64

typedef __attribute__((ext_vector_type(16))) __bf16 v16bf;
typedef __attribute__((ext_vector_type(8)))  float  v8f;

__device__ __forceinline__ v8f wmma_bf16(v16bf a, v16bf b, v8f c) {
  // D = A(16x32 bf16) * B(32x16 bf16) + C(16x16 f32)
  return __builtin_amdgcn_wmma_f32_16x16x32_bf16(false, a, false, b, (short)0, c,
                                                 false, false);
}
#define SCHED_FENCE() __builtin_amdgcn_sched_barrier(0)

// B-operand / Q-row load: 16 contiguous bf16 (32B) -> two b128 loads.
__device__ __forceinline__ v16bf ld16_contig(const __bf16* p) {
  union { v16bf v; uint4 q[2]; } u;
  u.q[0] = *(const uint4*)(p);
  u.q[1] = *(const uint4*)(p + 8);
  return u.v;
}
// A-operand load: 8 bf16 at p, 8 bf16 at p+8 (K = hi*16 + 0..15, contiguous).
__device__ __forceinline__ v16bf ld16_split(const __bf16* p) {
  union { v16bf v; uint4 q[2]; } u;
  u.q[0] = *(const uint4*)(p);
  u.q[1] = *(const uint4*)(p + 8);
  return u.v;
}
__device__ __forceinline__ unsigned pk2(float lo, float hi) {
  union { __bf16 h[2]; unsigned u; } t;
  t.h[0] = (__bf16)lo; t.h[1] = (__bf16)hi;
  return t.u;
}

// ---- transpose + convert weights: w[C,H] f32 -> wT[H,C] bf16 -----------------
__global__ void prep_w(const float* __restrict__ wq, const float* __restrict__ wk,
                       const float* __restrict__ wv, __bf16* __restrict__ wTq,
                       __bf16* __restrict__ wTk, __bf16* __restrict__ wTv) {
  int idx = blockIdx.x * 256 + threadIdx.x;        // over H*C
  if (idx >= HH * CC) return;
  int h = idx / CC, c = idx % CC;
  wTq[idx] = (__bf16)wq[c * HH + h];
  wTk[idx] = (__bf16)wk[c * HH + h];
  wTv[idx] = (__bf16)wv[c * HH + h];
}

// ---- projection helpers ------------------------------------------------------
__device__ __forceinline__ void load_a(const float* xrow, int k0, int hi,
                                       float4* f) {
  const float* xp = xrow + k0 + hi * 16;
  f[0] = *(const float4*)(xp);
  f[1] = *(const float4*)(xp + 4);
  f[2] = *(const float4*)(xp + 8);
  f[3] = *(const float4*)(xp + 12);
}
__device__ __forceinline__ v16bf cvtA(const float4* f) {
  v16bf a;
  a[0]=(__bf16)f[0].x;  a[1]=(__bf16)f[0].y;  a[2]=(__bf16)f[0].z;  a[3]=(__bf16)f[0].w;
  a[4]=(__bf16)f[1].x;  a[5]=(__bf16)f[1].y;  a[6]=(__bf16)f[1].z;  a[7]=(__bf16)f[1].w;
  a[8]=(__bf16)f[2].x;  a[9]=(__bf16)f[2].y;  a[10]=(__bf16)f[2].z; a[11]=(__bf16)f[2].w;
  a[12]=(__bf16)f[3].x; a[13]=(__bf16)f[3].y; a[14]=(__bf16)f[3].z; a[15]=(__bf16)f[3].w;
  return a;
}
__device__ __forceinline__ void load_b(const __bf16* __restrict__ wTq,
                                       const __bf16* __restrict__ wTk,
                                       const __bf16* __restrict__ wTv,
                                       int k0, int n, int hi,
                                       v16bf* bq, v16bf* bk, v16bf* bv) {
#pragma unroll
  for (int t = 0; t < 4; ++t) {
    size_t wo = (size_t)(t * 16 + n) * CC + k0 + hi * 16;
    bq[t] = ld16_contig(wTq + wo);
    bk[t] = ld16_contig(wTk + wo);
    bv[t] = ld16_contig(wTv + wo);
  }
}
__device__ __forceinline__ void mac12(v16bf a, const v16bf* bq, const v16bf* bk,
                                      const v16bf* bv, v8f* accq, v8f* acck,
                                      v8f* accv) {
#pragma unroll
  for (int t = 0; t < 4; ++t) {
    accq[t] = wmma_bf16(a, bq[t], accq[t]);
    acck[t] = wmma_bf16(a, bk[t], acck[t]);
    accv[t] = wmma_bf16(a, bv[t], accv[t]);
  }
}

// ---- q/k/v projection: [32768,1024] x [1024,64], software-pipelined ----------
__global__ __launch_bounds__(256) void proj_kernel(
    const float* __restrict__ x, const __bf16* __restrict__ wTq,
    const __bf16* __restrict__ wTk, const __bf16* __restrict__ wTv,
    __bf16* __restrict__ qout, __bf16* __restrict__ kout,
    __bf16* __restrict__ vTout) {
  const int lane = threadIdx.x & 31;
  const int wave = threadIdx.x >> 5;
  const int n = lane & 15, hi = lane >> 4;
  const int m0 = (blockIdx.x * 8 + wave) * 16;     // global row (b*T + t)

  v8f accq[4], acck[4], accv[4];
#pragma unroll
  for (int t = 0; t < 4; ++t) { accq[t]=(v8f){}; acck[t]=(v8f){}; accv[t]=(v8f){}; }

  const float* xrow = x + (size_t)(m0 + n) * CC;

  float4 F[2][4];
  v16bf Bq[2][4], Bk[2][4], Bv[2][4];

  load_a(xrow, 0, hi, F[0]);
  load_b(wTq, wTk, wTv, 0, n, hi, Bq[0], Bk[0], Bv[0]);

  for (int k0 = 0; k0 < CC; k0 += 64) {            // 16 iterations, 2 k-steps each
    // stage i+1 loads, fenced so the scheduler cannot sink them under the MACs
    load_a(xrow, k0 + 32, hi, F[1]);
    load_b(wTq, wTk, wTv, k0 + 32, n, hi, Bq[1], Bk[1], Bv[1]);
    SCHED_FENCE();
    mac12(cvtA(F[0]), Bq[0], Bk[0], Bv[0], accq, acck, accv);
    SCHED_FENCE();
    if (k0 + 64 < CC) {
      load_a(xrow, k0 + 64, hi, F[0]);
      load_b(wTq, wTk, wTv, k0 + 64, n, hi, Bq[0], Bk[0], Bv[0]);
    }
    SCHED_FENCE();
    mac12(cvtA(F[1]), Bq[1], Bk[1], Bv[1], accq, acck, accv);
    SCHED_FENCE();
  }

  // C/D layout: VGPR r holds (row = r + hi*8, col = n)
  const int bidx = m0 >> 11;                       // row / T
#pragma unroll
  for (int t = 0; t < 4; ++t) {
#pragma unroll
    for (int r = 0; r < 8; ++r) {
      int row = m0 + r + hi * 8;
      int h = t * 16 + n;
      qout[(size_t)row * HH + h] = (__bf16)accq[t][r];
      kout[(size_t)row * HH + h] = (__bf16)acck[t][r];
      int tloc = row & (TT - 1);
      vTout[((size_t)bidx * HH + h) * TT + tloc] = (__bf16)accv[t][r]; // store V^T
    }
  }
}

// ---- attention helpers -------------------------------------------------------
__device__ __forceinline__ void load_k4(const __bf16* __restrict__ kbb, int kv0,
                                        int n, int hi, v16bf* ka) {
  const __bf16* kp0 = kbb + (size_t)(kv0 + n) * HH + hi * 16;
  const __bf16* kp1 = kbb + (size_t)(kv0 + 16 + n) * HH + hi * 16;
  ka[0] = ld16_split(kp0);
  ka[1] = ld16_split(kp0 + 32);
  ka[2] = ld16_split(kp1);
  ka[3] = ld16_split(kp1 + 32);
}

// ---- flash attention, S^T orientation, pipelined -----------------------------
// One wave = 16 query rows.  S^T = K_tile * Q^T (softmax reduction mostly
// in-lane), O^T = V^T * P^T (P^T relayout via shfl_xor(16) only).
__global__ __launch_bounds__(256) void attn_kernel(
    const __bf16* __restrict__ qbf, const __bf16* __restrict__ kbf,
    const __bf16* __restrict__ vT, float* __restrict__ out) {
  const int lane = threadIdx.x & 31;
  const int wave = threadIdx.x >> 5;
  const int n = lane & 15, hi = lane >> 4;
  const int tile = blockIdx.x * 8 + wave;          // 0 .. B*T/16-1
  const int b = tile >> 7;                         // 128 tiles per batch
  const int m0 = (tile & 127) << 4;
  const size_t bt0 = (size_t)b * TT;
  const int m = m0 + n;                            // this lane's query row (local)
  const float sc = 0.125f * 1.44269504f;           // 1/sqrt(64) * log2(e)

  const __bf16* kbb = kbf + bt0 * HH;
  const __bf16* vbb = vT + ((size_t)b * HH + n) * TT + hi * 16;

  // Q^T B-operand: lane holds Q row (m0+n), K = c*32 + hi*16 + 0..15 (contig)
  v16bf qb0 = ld16_contig(qbf + (bt0 + m) * HH + 0  + hi * 16);
  v16bf qb1 = ld16_contig(qbf + (bt0 + m) * HH + 32 + hi * 16);

  float mstate = -1e30f, lstate = 0.0f;
  v8f o0 = {}, o1 = {}, o2 = {}, o3 = {};          // O^T tiles, h = t*16+r+hi*8, col m=n

  v16bf ka[4], kan[4];
  load_k4(kbb, 0, n, hi, ka);                      // prologue prefetch

  for (int kv0 = 0; kv0 <= m0 + 15; kv0 += 32) {
    // V loads issued first: consumed only after softmax, latency fully hidden
    v16bf va0 = ld16_split(vbb + (size_t)0  * TT + kv0);
    v16bf va1 = ld16_split(vbb + (size_t)16 * TT + kv0);
    v16bf va2 = ld16_split(vbb + (size_t)32 * TT + kv0);
    v16bf va3 = ld16_split(vbb + (size_t)48 * TT + kv0);
    SCHED_FENCE();

    // ----- S^T: two 16-kv tiles from prefetched K operands -----
    v8f s0 = {}, s1 = {};
    s0 = wmma_bf16(ka[0], qb0, s0);
    s0 = wmma_bf16(ka[1], qb1, s0);
    s1 = wmma_bf16(ka[2], qb0, s1);
    s1 = wmma_bf16(ka[3], qb1, s1);

    const bool more = (kv0 + 32 <= m0 + 15);       // wave-uniform
    if (more) load_k4(kbb, kv0 + 32, n, hi, kan);  // prefetch next chunk's K
    SCHED_FENCE();                                 // keep prefetch above softmax

    // scale(+log2e) + causal mask; element (kv = kv0[+16]+r+hi*8, query col m)
    float tmax = -1e30f;
#pragma unroll
    for (int r = 0; r < 8; ++r) {
      int kva = kv0 + r + hi * 8;
      float va = (kva <= m) ? s0[r] * sc : -1e30f;
      float vb = (kva + 16 <= m) ? s1[r] * sc : -1e30f;
      s0[r] = va; s1[r] = vb;
      tmax = fmaxf(tmax, fmaxf(va, vb));
    }
    tmax = fmaxf(tmax, __shfl_xor(tmax, 16, 32));  // combine kv halves (lane pair)

    float newm = fmaxf(mstate, tmax);
    float alpha = __builtin_amdgcn_exp2f(mstate - newm);
    float psum = 0.0f;
#pragma unroll
    for (int r = 0; r < 8; ++r) {
      float pa = __builtin_amdgcn_exp2f(s0[r] - newm);
      float pb = __builtin_amdgcn_exp2f(s1[r] - newm);
      s0[r] = pa; s1[r] = pb;
      psum += pa + pb;
    }
    psum += __shfl_xor(psum, 16, 32);
    lstate = lstate * alpha + psum;
    mstate = newm;
#pragma unroll
    for (int r = 0; r < 8; ++r) { o0[r]*=alpha; o1[r]*=alpha; o2[r]*=alpha; o3[r]*=alpha; }

    // ----- P^T (32kv x 16m) as WMMA B-operand: lane-pair swap via shfl_xor(16)
    unsigned p0[4], p1[4];
#pragma unroll
    for (int d = 0; d < 4; ++d) {
      p0[d] = pk2(s0[2*d], s0[2*d+1]);
      p1[d] = pk2(s1[2*d], s1[2*d+1]);
    }
    union { v16bf v; unsigned d[8]; } pb;
#pragma unroll
    for (int d = 0; d < 4; ++d) {
      unsigned x0 = __shfl_xor(p0[d], 16, 32);
      unsigned x1 = __shfl_xor(p1[d], 16, 32);
      pb.d[d]     = hi ? x1 : p0[d];   // K = hi*16 + 0..7
      pb.d[d + 4] = hi ? p1[d] : x0;   // K = hi*16 + 8..15
    }

    // ----- O^T += V^T * P^T; A = V^T rows (h per lane), contiguous kv chunks -
    o0 = wmma_bf16(va0, pb.v, o0);
    o1 = wmma_bf16(va1, pb.v, o1);
    o2 = wmma_bf16(va2, pb.v, o2);
    o3 = wmma_bf16(va3, pb.v, o3);

    if (more) {
#pragma unroll
      for (int i = 0; i < 4; ++i) ka[i] = kan[i];
    }
  }

  const float invl = 1.0f / lstate;                // lane-uniform per query col
  float* op = out + (bt0 + m) * HH;
#pragma unroll
  for (int r = 0; r < 8; ++r) {
    int hrow = r + hi * 8;
    op[hrow]      = o0[r] * invl;
    op[16 + hrow] = o1[r] * invl;
    op[32 + hrow] = o2[r] * invl;
    op[48 + hrow] = o3[r] * invl;
  }
}

extern "C" void kernel_launch(void* const* d_in, const int* in_sizes, int n_in,
                              void* d_out, int out_size, void* d_ws, size_t ws_size,
                              hipStream_t stream) {
  (void)in_sizes; (void)n_in; (void)out_size; (void)ws_size;
  const float* x  = (const float*)d_in[0];
  const float* wq = (const float*)d_in[1];
  const float* wk = (const float*)d_in[2];
  const float* wv = (const float*)d_in[3];
  float* out = (float*)d_out;

  // workspace: q,k (bf16 [B*T,H]), v^T (bf16 [B,H,T]), w^T x3 (bf16 [H,C]) ~13MB
  __bf16* qbf = (__bf16*)d_ws;
  __bf16* kbf = qbf + (size_t)BB * TT * HH;
  __bf16* vT  = kbf + (size_t)BB * TT * HH;
  __bf16* wTq = vT  + (size_t)BB * TT * HH;
  __bf16* wTk = wTq + (size_t)HH * CC;
  __bf16* wTv = wTk + (size_t)HH * CC;

  prep_w<<<(HH * CC + 255) / 256, 256, 0, stream>>>(wq, wk, wv, wTq, wTk, wTv);
  proj_kernel<<<(BB * TT) / (16 * 8), 256, 0, stream>>>(x, wTq, wTk, wTv,
                                                        qbf, kbf, vT);
  attn_kernel<<<(BB * TT / 16) / 8, 256, 0, stream>>>(qbf, kbf, vT, out);
}
